// Decoder_56289841382057
// MI455X (gfx1250) — matrix operand order repaired and checked by reading, not compiled, not run on hardware
//
#include <hip/hip_runtime.h>
#include <stdint.h>

typedef unsigned short u16;
typedef __attribute__((ext_vector_type(16))) __bf16 v16bf;
typedef __attribute__((ext_vector_type(8)))  float  v8f;

#define T_SEQ     2048
#define E_DIM     128
#define H_DIM     512
#define G3H       1536
#define DK        256
#define NKEYS     50000
#define NKEYS_PAD 50048
#define VOCAB     1704
#define VOCAB_PAD 1712
#define NLOC      100
#define NSPLIT    16
#define SPLIT_LEN 3125      /* 50000/16 */
#define NCHUNK    98        /* ceil(3125/32) */
#define MT        128

// Flip to 0 if the async-to-LDS mnemonic is rejected by the assembler.
#ifndef USE_ASYNC_LDS
#define USE_ASYNC_LDS 1
#endif

// ---------------- bf16 helpers ----------------
static __device__ __forceinline__ u16 f2bf(float f) {
  union { float f; unsigned int i; } u; u.f = f;
  unsigned int i = u.i;
  return (u16)((i + 0x7fffu + ((i >> 16) & 1u)) >> 16);   // RNE
}
static __device__ __forceinline__ float bf2f(u16 h) {
  union { unsigned int i; float f; } u; u.i = ((unsigned int)h) << 16;
  return u.f;
}

union FragU { uint4 q[2]; v16bf v; };

// A-matrix fragment (16xK tile, rows stored K-major):
// lane<16 : row=lane,    K=0..15 contiguous; lane>=16: row=lane-16, K=16..31
static __device__ __forceinline__ v16bf load_a32(const u16* rowk, int half) {
  FragU x;
  x.q[0] = *(const uint4*)(rowk + half * 16);
  x.q[1] = *(const uint4*)(rowk + half * 16 + 8);
  return x.v;
}
// B-matrix fragment (Kx16 tile) from K-major storage of B^T rows:
// lane<16 : col=lane, K=0..15 contiguous; lane>=16: col=lane-16, K=16..31
// (caller passes rowptr + k + half*16)
static __device__ __forceinline__ v16bf load_b32(const u16* rowk_half) {
  FragU x;
  x.q[0] = *(const uint4*)(rowk_half);
  x.q[1] = *(const uint4*)(rowk_half + 8);
  return x.v;
}

static __device__ __forceinline__ v8f wmma_bf16(v16bf a, v16bf b, v8f c) {
  return __builtin_amdgcn_wmma_f32_16x16x32_bf16(false, a, false, b, (short)0, c,
                                                 false, false);
}

static __device__ __forceinline__ void wait_asynccnt0() {
#if __has_builtin(__builtin_amdgcn_s_wait_asynccnt)
  __builtin_amdgcn_s_wait_asynccnt(0);
#else
  asm volatile("s_wait_asynccnt 0x0" ::: "memory");
#endif
}

// Per-lane async global->LDS copy of 16 bytes (GLOBAL_LOAD_ASYNC_TO_LDS_B128).
// ldsaddr: LDS byte address (low 32 bits of the generic shared pointer;
// hardware maps LDS flat addresses via addr[31:0]).
static __device__ __forceinline__ void async_copy_b128(unsigned ldsaddr,
                                                       const void* gaddr) {
  asm volatile("global_load_async_to_lds_b128 %0, %1, off"
               :: "v"(ldsaddr), "v"(gaddr) : "memory");
}

// ---------------- small utility kernels ----------------
__global__ void k_embed(const int* __restrict__ X, const float* __restrict__ emb,
                        float* __restrict__ y, u16* __restrict__ ybf) {
  int idx = blockIdx.x * blockDim.x + threadIdx.x;
  if (idx >= T_SEQ * E_DIM) return;
  int t = idx >> 7, e = idx & 127;
  float v = emb[X[t] * E_DIM + e];
  y[idx] = v; ybf[idx] = f2bf(v);
}

__global__ void k_cvt(const float* __restrict__ src, u16* __restrict__ dst,
                      long long nsrc, long long ndst) {
  long long i = (long long)blockIdx.x * blockDim.x + threadIdx.x;
  if (i >= ndst) return;
  dst[i] = (i < nsrc) ? f2bf(src[i]) : (u16)0;
}

__global__ void k_cvt_t(const float* __restrict__ src, u16* __restrict__ dst,
                        int R, int C) {   // src (R,C) f32 -> dst (C,R) bf16
  int i = blockIdx.x * blockDim.x + threadIdx.x;
  if (i >= R * C) return;
  int r = i / C, c = i % C;
  dst[c * R + r] = f2bf(src[i]);
}

__global__ void k_cvt_pad2d(const float* __restrict__ src, u16* __restrict__ dst,
                            int R, int C, int Rp, int Cp) {
  int i = blockIdx.x * blockDim.x + threadIdx.x;
  if (i >= Rp * Cp) return;
  int r = i / Cp, c = i % Cp;
  dst[i] = (r < R && c < C) ? f2bf(src[r * C + c]) : (u16)0;
}

// ---------------- generic NT GEMM: C(MxN) = A(MxK,bf16) * B(NxK,bf16)^T + bias ----
// B tile (16 x K, K<=512) staged once per block in LDS (async copy on CDNA5),
// shared by all 8 waves; A streamed from global.
__global__ __launch_bounds__(256) void k_gemm_nt(
    const u16* __restrict__ A, int lda, const u16* __restrict__ B, int ldb,
    float* __restrict__ C, int ldc, const float* __restrict__ bias, int N, int K) {
  __shared__ __align__(16) u16 Btile[16][512];   // 16 KB
  int wave = threadIdx.x >> 5, lane = threadIdx.x & 31;
  int half = lane >> 4, sub = lane & 15;
  int m0 = blockIdx.x * 128 + wave * 16;
  int n0 = blockIdx.y * 16;

  // stage B tile: 16 rows x K bf16 = 2*K chunks of 8 elements (16B)
  int kc = K >> 3;                 // chunks per row (power of two: 16 or 64)
  int ksh = 31 - __clz(kc);
  for (int it = threadIdx.x; it < 2 * K; it += 256) {
    int r = it >> ksh;
    int c8 = (it & (kc - 1)) * 8;
    const u16* gp = B + (size_t)(n0 + r) * ldb + c8;
#if USE_ASYNC_LDS
    async_copy_b128((unsigned)(uintptr_t)&Btile[r][c8], (const void*)gp);
#else
    *(uint4*)&Btile[r][c8] = *(const uint4*)gp;
#endif
  }
#if USE_ASYNC_LDS
  wait_asynccnt0();
#endif
  __syncthreads();

  const u16* arow = A + (size_t)(m0 + sub) * lda;
  v8f acc = {};
  for (int k = 0; k < K; k += 32) {
    v16bf a = load_a32(arow + k, half);
    v16bf b = load_b32(&Btile[sub][k + half * 16]);
    acc = wmma_bf16(a, b, acc);
  }
  int n = n0 + sub;
  if (n < N) {
    float bv = bias ? bias[n] : 0.0f;
    int m = m0 + 8 * half;
#pragma unroll
    for (int r = 0; r < 8; ++r) C[(size_t)(m + r) * ldc + n] = acc[r] + bv;
  }
}

// ---------------- GRU sequential scan: 1 block, 512 threads, h in LDS ----------------
__global__ __launch_bounds__(512) void k_gru(
    const float* __restrict__ gx, const u16* __restrict__ Whh,
    const float* __restrict__ b_hh, const float* __restrict__ glo0,
    const float* __restrict__ loch0, float* __restrict__ out,
    u16* __restrict__ outbf) {
  __shared__ __align__(16) float h[H_DIM];
  int j = threadIdx.x;
  h[j] = (j < 256) ? glo0[j] : loch0[j - 256];
  __syncthreads();
  const u16* w0 = Whh + (size_t)j * H_DIM;
  const u16* w1 = Whh + (size_t)(j + H_DIM) * H_DIM;
  const u16* w2 = Whh + (size_t)(j + 2 * H_DIM) * H_DIM;
  float bh0 = b_hh[j], bh1 = b_hh[j + H_DIM], bh2 = b_hh[j + 2 * H_DIM];
  for (int t = 0; t < T_SEQ; ++t) {
    float g0 = bh0, g1 = bh1, g2 = bh2;
    for (int k = 0; k < H_DIM; k += 8) {
      float4 ha = *(const float4*)&h[k];
      float4 hb = *(const float4*)&h[k + 4];
      uint4 a0 = *(const uint4*)(w0 + k);
      uint4 a1 = *(const uint4*)(w1 + k);
      uint4 a2 = *(const uint4*)(w2 + k);
      const u16* p0 = (const u16*)&a0;
      const u16* p1 = (const u16*)&a1;
      const u16* p2 = (const u16*)&a2;
      float hv[8] = {ha.x, ha.y, ha.z, ha.w, hb.x, hb.y, hb.z, hb.w};
#pragma unroll
      for (int i = 0; i < 8; ++i) {
        g0 = fmaf(hv[i], bf2f(p0[i]), g0);
        g1 = fmaf(hv[i], bf2f(p1[i]), g1);
        g2 = fmaf(hv[i], bf2f(p2[i]), g2);
      }
    }
    const float* gxt = gx + (size_t)t * G3H;
    float r = 1.0f / (1.0f + __expf(-(gxt[j] + g0)));
    float z = 1.0f / (1.0f + __expf(-(gxt[j + H_DIM] + g1)));
    float n = tanhf(gxt[j + 2 * H_DIM] + r * g2);
    float hp = h[j];
    float hn = (1.0f - z) * n + z * hp;
    __syncthreads();
    h[j] = hn;
    out[(size_t)t * H_DIM + j] = hn;
    outbf[(size_t)t * H_DIM + j] = f2bf(hn);
    __syncthreads();
  }
}

// ---------------- flash attention over 50000 keys (split-K, WMMA) ----------------
// grid (16 m-tiles, 16 n-splits), 8 waves; partial (ctx,l) per split
__global__ __launch_bounds__(256) void k_flash(
    const u16* __restrict__ Q, const u16* __restrict__ KV,
    float* __restrict__ pctx, float* __restrict__ pl) {
  __shared__ __align__(16) u16 Vt[DK][32];      // V chunk transposed (e, key): 16 KB
  __shared__ __align__(16) u16 Pl[8][16][32];   // per-wave exp weights (q, key): 8 KB
  int wave = threadIdx.x >> 5, lane = threadIdx.x & 31;
  int half = lane >> 4, sub = lane & 15;
  int mt = blockIdx.x, ns = blockIdx.y;
  int qrow = mt * MT + wave * 16 + sub;
  const u16* qp = Q + (size_t)qrow * DK;
  v16bf bq[8];
#pragma unroll
  for (int f = 0; f < 8; ++f) bq[f] = load_b32(qp + f * 32 + half * 16);
  v8f zero = {};
  v8f ctx[16];
#pragma unroll
  for (int i = 0; i < 16; ++i) ctx[i] = zero;
  float lsum = 0.0f;
  int base = ns * SPLIT_LEN;
  int limit = base + SPLIT_LEN;
  for (int cc = 0; cc < NCHUNK; ++cc) {
    int k0 = base + cc * 32;
    __syncthreads();
    // stage V^T tile (zero OOB rows so WMMA never multiplies poison into ctx)
    for (int it = threadIdx.x; it < 32 * 32; it += 256) {
      int row = it >> 5;
      int cg = (it & 31) * 8;
      int key = k0 + row;
      uint4 d; d.x = 0u; d.y = 0u; d.z = 0u; d.w = 0u;
      if (key < limit) d = *(const uint4*)(KV + (size_t)key * DK + cg);
      const u16* d8 = (const u16*)&d;
#pragma unroll
      for (int i = 0; i < 8; ++i) Vt[cg + i][row] = d8[i];
    }
    if (cc + 1 < NCHUNK) {   // global_prefetch_b8 of next 16 KB key chunk
      const char* pf = (const char*)(KV + (size_t)(k0 + 32) * DK) +
                       (size_t)threadIdx.x * 64;
      __builtin_prefetch(pf, 0, 0);
    }
    __syncthreads();
    // scores: keys as A-matrix, queries as B-matrix; exp weights -> LDS (re-layout)
#pragma unroll
    for (int kb = 0; kb < 2; ++kb) {
      int keyrow = k0 + kb * 16 + sub;
      const u16* kp = KV + (size_t)keyrow * DK;
      v8f s = zero;
#pragma unroll
      for (int f = 0; f < 8; ++f) {
        v16bf ak = load_a32(kp + f * 32, half);
        s = wmma_bf16(ak, bq[f], s);
      }
      int kbase = kb * 16 + 8 * half;
#pragma unroll
      for (int r = 0; r < 8; ++r) {
        int key = k0 + kbase + r;
        float e = (key < limit) ? __expf(s[r]) : 0.0f;  // bounded scores: no max needed
        lsum += e;                                       // per-lane: query 'sub'
        Pl[wave][sub][kbase + r] = f2bf(e);
      }
    }
    // ctx(16q x 16e chunks) += P(16q x 32k) @ V(32k x 16e)
    v16bf ap = load_a32(&Pl[wave][sub][0], half);
#pragma unroll
    for (int ec = 0; ec < 16; ++ec) {
      v16bf bv = load_b32(&Vt[ec * 16 + sub][0] + half * 16);
      ctx[ec] = wmma_bf16(ap, bv, ctx[ec]);
    }
  }
  lsum += __shfl_xor(lsum, 16, 32);   // combine key halves for query 'sub'
  int po = (mt * NSPLIT + ns) * MT + wave * 16;
  if (half == 0) pl[po + sub] = lsum;
  int rowb = po + 8 * half;
#pragma unroll
  for (int ec = 0; ec < 16; ++ec)
#pragma unroll
    for (int r = 0; r < 8; ++r)
      pctx[(size_t)(rowb + r) * DK + ec * 16 + sub] = ctx[ec][r];
}

__global__ void k_combine(const float* __restrict__ pctx, const float* __restrict__ pl,
                          float* __restrict__ cg) {
  int idx = blockIdx.x * blockDim.x + threadIdx.x;
  if (idx >= T_SEQ * DK) return;
  int m = idx >> 8, e = idx & 255;
  int mtile = m >> 7, mr = m & 127;
  float num = 0.0f, den = 0.0f;
  for (int s = 0; s < NSPLIT; ++s) {
    num += pctx[(size_t)((mtile * NSPLIT + s) * MT + mr) * DK + e];
    den += pl[(mtile * NSPLIT + s) * MT + mr];
  }
  cg[idx] = num / den;
}

// ---------------- local attention (100 keys) ----------------
__global__ __launch_bounds__(128) void k_local(
    const float* __restrict__ ql, const float* __restrict__ loc,
    float* __restrict__ cl, float* __restrict__ beta) {
  __shared__ float p[128];
  __shared__ float red[128];
  int t = blockIdx.x, tid = threadIdx.x;
  const float* q = ql + (size_t)t * DK;
  float s = -3.0e38f;
  if (tid < NLOC) {
    const float* g = loc + (size_t)tid * 2048;  // loc_enc_outputs[:,0,:]
    float a = 0.0f;
    for (int e = 0; e < DK; ++e) a += g[e] * q[e];
    s = a;
  }
  red[tid] = s; __syncthreads();
  for (int off = 64; off > 0; off >>= 1) {
    if (tid < off) red[tid] = fmaxf(red[tid], red[tid + off]);
    __syncthreads();
  }
  float mx = red[0]; __syncthreads();
  float e = (tid < NLOC) ? __expf(s - mx) : 0.0f;
  red[tid] = e; __syncthreads();
  for (int off = 64; off > 0; off >>= 1) {
    if (tid < off) red[tid] += red[tid + off];
    __syncthreads();
  }
  float pv = e / red[0];
  p[tid] = pv;
  if (tid < NLOC) beta[(size_t)t * NLOC + tid] = pv;
  __syncthreads();
  for (int e2 = tid; e2 < DK; e2 += 128) {
    float a = 0.0f;
    for (int n = 0; n < NLOC; ++n) a += p[n] * loc[(size_t)n * 2048 + e2];
    cl[(size_t)t * DK + e2] = a;
  }
}

// ---------------- head: P_vocab softmax, p_gen, P_w (K padded to 128, bf16) ----------
__global__ __launch_bounds__(128) void k_head(
    const float* __restrict__ out, const float* __restrict__ cl,
    const float* __restrict__ cg, const float* __restrict__ y,
    const float* __restrict__ Wv, const float* __restrict__ Wvb,
    const float* __restrict__ wh, const float* __restrict__ wc,
    const float* __restrict__ wy, const float* __restrict__ beta,
    u16* __restrict__ Pwbf) {
  __shared__ float red[128];
  __shared__ float s_pg;
  int t = blockIdx.x, tid = threadIdx.x;
  const float* o  = out + (size_t)t * H_DIM;
  const float* cv = cl + (size_t)t * DK;
  const float* gv = cg + (size_t)t * DK;
  float logit = -3.0e38f;
  if (tid < NLOC) {
    const float* w = Wv + (size_t)tid * 1024;
    float a = Wvb[tid];
    for (int k = 0; k < H_DIM; ++k) a += o[k] * w[k];
    for (int k = 0; k < DK; ++k) a += cv[k] * w[H_DIM + k];
    for (int k = 0; k < DK; ++k) a += gv[k] * w[H_DIM + DK + k];
    logit = a;
  }
  red[tid] = logit; __syncthreads();
  for (int off = 64; off > 0; off >>= 1) {
    if (tid < off) red[tid] = fmaxf(red[tid], red[tid + off]);
    __syncthreads();
  }
  float mx = red[0]; __syncthreads();
  float e = (tid < NLOC) ? __expf(logit - mx) : 0.0f;
  red[tid] = e; __syncthreads();
  for (int off = 64; off > 0; off >>= 1) {
    if (tid < off) red[tid] += red[tid + off];
    __syncthreads();
  }
  float inv = 1.0f / red[0];
  if (tid == 0) {
    const float* yv = y + (size_t)t * E_DIM;
    float a = 0.0f;
    for (int k = 0; k < H_DIM; ++k) a += o[k] * wh[k];
    for (int k = 0; k < DK; ++k) a += cv[k] * wc[k];
    for (int k = 0; k < E_DIM; ++k) a += yv[k] * wy[k];
    s_pg = (a >= 0.0f) ? a : 0.2f * a;   // leaky_relu 0.2
  }
  __syncthreads();
  float pg = s_pg;
  float pw = 0.0f;
  if (tid < NLOC) pw = pg * (e * inv) + (1.0f - pg) * beta[(size_t)t * NLOC + tid];
  Pwbf[(size_t)t * 128 + tid] = f2bf(pw);   // K-pad cols 100..127 = 0
}

// ---------------- host launcher ----------------
extern "C" void kernel_launch(void* const* d_in, const int* in_sizes, int n_in,
                              void* d_out, int out_size, void* d_ws, size_t ws_size,
                              hipStream_t stream) {
  (void)in_sizes; (void)n_in; (void)out_size; (void)ws_size;
  const int*   X      = (const int*)d_in[0];
  const float* glo    = (const float*)d_in[1];
  const float* lochid = (const float*)d_in[2];
  const float* locout = (const float*)d_in[3];
  const float* emb    = (const float*)d_in[4];
  const float* W_ih   = (const float*)d_in[5];
  const float* W_hh   = (const float*)d_in[6];
  const float* b_ih   = (const float*)d_in[7];
  const float* b_hh   = (const float*)d_in[8];
  const float* W_ga_g = (const float*)d_in[9];
  const float* W_ga_l = (const float*)d_in[10];
  const float* Wv_w   = (const float*)d_in[11];
  const float* Wv_b   = (const float*)d_in[12];
  const float* whp    = (const float*)d_in[13];
  const float* wcp    = (const float*)d_in[14];
  const float* wyp    = (const float*)d_in[15];
  const float* fc_w   = (const float*)d_in[16];
  const float* fc_b   = (const float*)d_in[17];
  float* outp = (float*)d_out;

  char* cur = (char*)d_ws;
  auto alloc = [&](size_t b) -> void* {
    void* r = (void*)cur;
    cur += (b + 255) & ~(size_t)255;
    return r;
  };
  float* y     = (float*)alloc(sizeof(float) * T_SEQ * E_DIM);
  u16*   ybf   = (u16*)  alloc(sizeof(u16) * T_SEQ * E_DIM);
  float* gx    = (float*)alloc(sizeof(float) * T_SEQ * G3H);
  float* outh  = (float*)alloc(sizeof(float) * T_SEQ * H_DIM);
  u16*   outbf = (u16*)  alloc(sizeof(u16) * T_SEQ * H_DIM);
  u16*   Wihbf = (u16*)  alloc(sizeof(u16) * G3H * E_DIM);
  u16*   Whhbf = (u16*)  alloc(sizeof(u16) * G3H * H_DIM);
  u16*   WgTg  = (u16*)  alloc(sizeof(u16) * DK * H_DIM);
  u16*   WgTl  = (u16*)  alloc(sizeof(u16) * DK * H_DIM);
  u16*   glob  = (u16*)  alloc(sizeof(u16) * NKEYS_PAD * DK);
  float* qg    = (float*)alloc(sizeof(float) * T_SEQ * DK);
  u16*   qgbf  = (u16*)  alloc(sizeof(u16) * T_SEQ * DK);
  float* qlv   = (float*)alloc(sizeof(float) * T_SEQ * DK);
  float* pctx  = (float*)alloc(sizeof(float) * 16 * NSPLIT * MT * DK);
  float* pl    = (float*)alloc(sizeof(float) * 16 * NSPLIT * MT);
  float* cgv   = (float*)alloc(sizeof(float) * T_SEQ * DK);
  float* clv   = (float*)alloc(sizeof(float) * T_SEQ * DK);
  float* beta  = (float*)alloc(sizeof(float) * T_SEQ * NLOC);
  u16*   Pwbf  = (u16*)  alloc(sizeof(u16) * T_SEQ * 128);
  u16*   fcwbf = (u16*)  alloc(sizeof(u16) * VOCAB_PAD * 128);

  k_embed<<<(T_SEQ * E_DIM + 255) / 256, 256, 0, stream>>>(X, emb, y, ybf);
  k_cvt<<<(G3H * E_DIM + 255) / 256, 256, 0, stream>>>(
      W_ih, Wihbf, (long long)G3H * E_DIM, (long long)G3H * E_DIM);
  k_cvt<<<(G3H * H_DIM + 255) / 256, 256, 0, stream>>>(
      W_hh, Whhbf, (long long)G3H * H_DIM, (long long)G3H * H_DIM);
  k_cvt<<<(NKEYS_PAD * DK + 255) / 256, 256, 0, stream>>>(
      glo, glob, (long long)NKEYS * DK, (long long)NKEYS_PAD * DK);
  k_cvt_t<<<(H_DIM * DK + 255) / 256, 256, 0, stream>>>(W_ga_g, WgTg, H_DIM, DK);
  k_cvt_t<<<(H_DIM * DK + 255) / 256, 256, 0, stream>>>(W_ga_l, WgTl, H_DIM, DK);
  k_cvt_pad2d<<<(VOCAB_PAD * 128 + 255) / 256, 256, 0, stream>>>(
      fc_w, fcwbf, VOCAB, NLOC, VOCAB_PAD, 128);

  // gx = y @ W_ih^T + b_ih   (2048x1536, K=128)
  k_gemm_nt<<<dim3(T_SEQ / 128, G3H / 16), 256, 0, stream>>>(
      ybf, E_DIM, Wihbf, E_DIM, gx, G3H, b_ih, G3H, E_DIM);
  // sequential GRU scan
  k_gru<<<1, H_DIM, 0, stream>>>(gx, Whhbf, b_hh, glo, lochid, outh, outbf);
  // q = out @ W_ga  (global + local), 2048x256, K=512
  k_gemm_nt<<<dim3(T_SEQ / 128, DK / 16), 256, 0, stream>>>(
      outbf, H_DIM, WgTg, H_DIM, qg, DK, (const float*)nullptr, DK, H_DIM);
  k_gemm_nt<<<dim3(T_SEQ / 128, DK / 16), 256, 0, stream>>>(
      outbf, H_DIM, WgTl, H_DIM, qlv, DK, (const float*)nullptr, DK, H_DIM);
  k_cvt<<<(T_SEQ * DK + 255) / 256, 256, 0, stream>>>(
      qg, qgbf, (long long)T_SEQ * DK, (long long)T_SEQ * DK);
  // global attention (flash split-K) + combine
  k_flash<<<dim3(T_SEQ / MT, NSPLIT), 256, 0, stream>>>(qgbf, glob, pctx, pl);
  k_combine<<<(T_SEQ * DK + 255) / 256, 256, 0, stream>>>(pctx, pl, cgv);
  // local attention + head
  k_local<<<T_SEQ, 128, 0, stream>>>(qlv, locout, clv, beta);
  k_head<<<T_SEQ, 128, 0, stream>>>(outh, clv, cgv, y, Wv_w, Wv_b, whp, wcp, wyp,
                                    beta, Pwbf);
  // logits = P_w @ fc_w^T + fc_b  (2048x1704, K padded to 128)
  k_gemm_nt<<<dim3(T_SEQ / 128, (VOCAB + 15) / 16), 256, 0, stream>>>(
      Pwbf, 128, fcwbf, 128, outp, VOCAB, fc_b, VOCAB, 128);
}
